// BeforeRNNAttention_21182778704866
// MI455X (gfx1250) — matrix operand order repaired and checked by reading, not compiled, not run on hardware
//
#include <hip/hip_runtime.h>
#include <hip/hip_bf16.h>
#include <math.h>

// Shapes from the reference
#define ESL   4096
#define BATCH 64
#define EHS   256
#define DHS   256

#define NC          32                 // number of s-chunks (kernel-1 grid.x)
#define ROWS_PER_WG (ESL / NC)         // 128 rows of s per workgroup
#define WAVES       8                  // 256 threads / wave32
#define TILE_R      16                 // rows per wave tile (WMMA M)
#define ROWP        (EHS + 8)          // padded LDS row stride (halves) to avoid bank conflicts
#define REC         (EHS + 2)          // ws record: m, Z, ctx[256]

typedef __attribute__((ext_vector_type(16))) _Float16 v16h;
typedef __attribute__((ext_vector_type(8)))  _Float16 h8;
typedef __attribute__((ext_vector_type(8)))  float    v8f;
typedef __attribute__((ext_vector_type(4)))  float    f4;

__global__ __launch_bounds__(256)
void attn_partial_kernel(const float* __restrict__ si, const float* __restrict__ h,
                         const float* __restrict__ W, const float* __restrict__ bias,
                         float* __restrict__ ws)
{
    __shared__ __align__(16) _Float16 tile[WAVES][TILE_R * ROWP]; // ~66 KB
    __shared__ __align__(16) _Float16 we16[EHS];                  // 512 B
    __shared__ float ctxbuf[WAVES][EHS];                          // 8 KB
    __shared__ float wstate[WAVES][2];
    __shared__ float red[WAVES];
    __shared__ float edec_s;

    const int t    = threadIdx.x;
    const int lane = t & 31;
    const int w    = t >> 5;
    const int c    = blockIdx.x;   // s-chunk
    const int b    = blockIdx.y;   // batch column

    // ---- init: We -> f16 in LDS, e_dec = si_1[b]·Wd + bias (block reduction) ----
    we16[t] = (_Float16)W[DHS + t];
    float prod = si[b * DHS + t] * W[t];
    #pragma unroll
    for (int off = 16; off; off >>= 1) prod += __shfl_xor(prod, off, 32);
    if (lane == 0) red[w] = prod;
    __syncthreads();
    if (t == 0) {
        float s = 0.f;
        #pragma unroll
        for (int i = 0; i < WAVES; ++i) s += red[i];
        edec_s = s + bias[0];
    }
    __syncthreads();
    const float edec = edec_s;

    // ---- wave-local online-softmax state ----
    float m = -INFINITY, Z = 0.f;
    float ctx[8] = {0.f, 0.f, 0.f, 0.f, 0.f, 0.f, 0.f, 0.f}; // column l = lane*8 + j

    _Float16* tp     = &tile[w][0];
    const int  hsel  = (lane >= 16) ? 8 : 0;  // A-fragment K half-select
    const int  r16   = lane & 15;

    for (int ti = w; ti < ROWS_PER_WG / TILE_R; ti += WAVES) {
        const int s0 = c * ROWS_PER_WG + ti * TILE_R;

        // Phase 1: coalesced global load of 16x256 f32 tile, convert to f16, stage in LDS
        #pragma unroll 4
        for (int r = 0; r < TILE_R; ++r) {
            const float* rp = h + ((size_t)(s0 + r) * BATCH + b) * EHS + lane * 8;
            f4 x0 = *(const f4*)rp;
            f4 x1 = *(const f4*)(rp + 4);
            h8 hv;
            hv[0] = (_Float16)x0[0]; hv[1] = (_Float16)x0[1];
            hv[2] = (_Float16)x0[2]; hv[3] = (_Float16)x0[3];
            hv[4] = (_Float16)x1[0]; hv[5] = (_Float16)x1[1];
            hv[6] = (_Float16)x1[2]; hv[7] = (_Float16)x1[3];
            *(h8*)(tp + r * ROWP + lane * 8) = hv;
        }

        // Phase 2: e_enc for 16 rows via WMMA (B = We broadcast over all 16 N columns)
        v8f cfrag = {0.f, 0.f, 0.f, 0.f, 0.f, 0.f, 0.f, 0.f};
        #pragma unroll
        for (int k0 = 0; k0 < EHS; k0 += 32) {
            // A: 16x32 f16, lane<16 holds row r16 K{k0..k0+7, k0+16..k0+23}; lane>=16 K{+8,+24}
            h8 a_lo = *(const h8*)(tp + r16 * ROWP + k0 + hsel);
            h8 a_hi = *(const h8*)(tp + r16 * ROWP + k0 + 16 + hsel);
            v16h a = __builtin_shufflevector(a_lo, a_hi,
                       0, 1, 2, 3, 4, 5, 6, 7, 8, 9, 10, 11, 12, 13, 14, 15);
            // B: 32x16 f16; lane<16 holds K{k0..k0+7, k0+16..k0+23}; lane>=16 K{+8,+24} (bcast over N)
            h8 b_lo = *(const h8*)(we16 + k0 + hsel);
            h8 b_hi = *(const h8*)(we16 + k0 + 16 + hsel);
            v16h bb = __builtin_shufflevector(b_lo, b_hi,
                       0, 1, 2, 3, 4, 5, 6, 7, 8, 9, 10, 11, 12, 13, 14, 15);
            cfrag = __builtin_amdgcn_wmma_f32_16x16x32_f16(
                        false, a, false, bb, (short)0, cfrag, false, false);
        }

        // Phase 3: energies (all N columns identical; rows 0-7 on lane0, 8-15 on lane16)
        float en[TILE_R];
        float mt = -INFINITY;
        #pragma unroll
        for (int r = 0; r < TILE_R; ++r) {
            float ev = __shfl(cfrag[r & 7], (r >> 3) * 16, 32);
            float e  = edec + ev;
            e = e > 0.f ? e : 0.f;           // relu before softmax
            en[r] = e;
            mt = fmaxf(mt, e);
        }

        // Phase 4: online-softmax rescale + ctx accumulation from the f16 LDS tile
        float mn = fmaxf(m, mt);
        float sc = __expf(m - mn);           // 0 on first tile (m = -inf)
        Z *= sc;
        #pragma unroll
        for (int j = 0; j < 8; ++j) ctx[j] *= sc;
        #pragma unroll
        for (int r = 0; r < TILE_R; ++r) {
            float p = __expf(en[r] - mn);
            Z += p;
            h8 hv = *(const h8*)(tp + r * ROWP + lane * 8);
            #pragma unroll
            for (int j = 0; j < 8; ++j) ctx[j] += p * (float)hv[j];
        }
        m = mn;
    }

    // ---- merge the 8 wave states inside the workgroup ----
    if (lane == 0) { wstate[w][0] = m; wstate[w][1] = Z; }
    #pragma unroll
    for (int j = 0; j < 8; ++j) ctxbuf[w][lane * 8 + j] = ctx[j];
    __syncthreads();

    float mg = -INFINITY;
    #pragma unroll
    for (int i = 0; i < WAVES; ++i) mg = fmaxf(mg, wstate[i][0]);
    float Zg = 0.f, acc = 0.f;
    #pragma unroll
    for (int i = 0; i < WAVES; ++i) {
        float s = __expf(wstate[i][0] - mg);
        Zg  += s * wstate[i][1];
        acc += s * ctxbuf[i][t];
    }

    float* rec = ws + ((size_t)(c * BATCH + b)) * REC;
    if (t == 0) { rec[0] = mg; rec[1] = Zg; }
    rec[2 + t] = acc;
}

__global__ __launch_bounds__(256)
void attn_combine_kernel(const float* __restrict__ ws, float* __restrict__ out)
{
    const int b = blockIdx.x;
    const int t = threadIdx.x;

    float mg = -INFINITY;
    for (int c = 0; c < NC; ++c)
        mg = fmaxf(mg, ws[((size_t)(c * BATCH + b)) * REC]);

    float Zg = 0.f, acc = 0.f;
    for (int c = 0; c < NC; ++c) {
        const float* rec = ws + ((size_t)(c * BATCH + b)) * REC;
        float s = __expf(rec[0] - mg);
        Zg  += s * rec[1];
        acc += s * rec[2 + t];
    }
    out[b * EHS + t] = acc / Zg;   // context[0, b, t]
}

extern "C" void kernel_launch(void* const* d_in, const int* in_sizes, int n_in,
                              void* d_out, int out_size, void* d_ws, size_t ws_size,
                              hipStream_t stream) {
    const float* si  = (const float*)d_in[0];  // (1, 64, 256)
    const float* h   = (const float*)d_in[1];  // (4096, 64, 256)
    const float* W   = (const float*)d_in[2];  // (1, 512)
    const float* bs  = (const float*)d_in[3];  // (1,)
    float* out = (float*)d_out;                // (1, 64, 256)
    float* ws  = (float*)d_ws;                 // NC*BATCH*258 floats = ~2.1 MB

    dim3 g1(NC, BATCH);
    attn_partial_kernel<<<g1, 256, 0, stream>>>(si, h, W, bs, ws);
    attn_combine_kernel<<<BATCH, 256, 0, stream>>>(ws, out);
}
